// AttentiveFPModel_51565377356339
// MI455X (gfx1250) — matrix operand order repaired and checked by reading, not compiled, not run on hardware
//
#include <hip/hip_runtime.h>

#define H 128
#define DIN 64
#define NLAYERS 3
#define TSTEPS 2

typedef __attribute__((ext_vector_type(16))) _Float16 v16h;
typedef __attribute__((ext_vector_type(2)))  _Float16 v2h;
typedef __attribute__((ext_vector_type(8)))  float    v8f;

// ---------- order-preserving float<->uint for atomic max ----------
__device__ __forceinline__ unsigned f2o(float f) {
  unsigned u = __float_as_uint(f);
  return (u & 0x80000000u) ? ~u : (u | 0x80000000u);
}
__device__ __forceinline__ float o2f(unsigned u) {
  return __uint_as_float((u & 0x80000000u) ? (u & 0x7fffffffu) : ~u);
}

// =====================================================================
// B-fragment packer: weight matrix -> f16 WMMA B-fragments.
//   transB==0 : B is [K,Ncols] row-major
//   transB==1 : B is [Ncols,K] row-major (multiply by B^T)
// Packed layout (dword units): (((ntile*KS + ks)*32 + lane)*8 + dw)
//   lane holds col = ntile*16 + (lane&15); K = ks*32 + (lane>>4)*16 + i.
// =====================================================================
__global__ void pack_b_kernel(const float* __restrict__ B, _Float16* __restrict__ Bp,
                              int K, int Ncols, int transB) {
  int tid = blockIdx.x * blockDim.x + threadIdx.x;
  int KS = K >> 5;
  int total = (Ncols >> 4) * KS * 256;      // dwords
  if (tid >= total) return;
  int dw    = tid & 7;
  int lane  = (tid >> 3) & 31;
  int rest  = tid >> 8;
  int ks    = rest % KS;
  int ntile = rest / KS;
  int col   = (ntile << 4) + (lane & 15);
  int kbase = (ks << 5) + ((lane >> 4) << 4) + (dw << 1);
  float v0, v1;
  if (transB) {
    v0 = B[(size_t)col * K + kbase];
    v1 = B[(size_t)col * K + kbase + 1];
  } else {
    v0 = B[(size_t)kbase * Ncols + col];
    v1 = B[(size_t)(kbase + 1) * Ncols + col];
  }
  v2h p; p[0] = (_Float16)v0; p[1] = (_Float16)v1;
  ((v2h*)Bp)[tid] = p;
}

// =====================================================================
// WMMA GEMM with pre-packed B: C[M,Ncols] = act(A[M,K] @ B + bias)
// One 16x16 tile per wave, 4 waves/block. K templated (fully unrolled).
// A loads: 4x b128 per k-step (row-clamped, branch-free: garbage rows only
// feed D rows that are never stored). B loads: contiguous v16h fragment.
// act: 0=none 1=relu 2=tanh
// =====================================================================
template <int K>
__global__ void wmma_gemm_packed_kernel(const float* __restrict__ A,
                                        const _Float16* __restrict__ Bp,
                                        const float* __restrict__ bias,
                                        float* __restrict__ C,
                                        int M, int Ncols, int act) {
  constexpr int KS = K >> 5;
  const int lane  = threadIdx.x & 31;
  const int wave  = threadIdx.x >> 5;
  const int ntile = blockIdx.y * 4 + wave;
  const int n0    = ntile << 4;
  if (n0 >= Ncols) return;                  // wave-uniform
  const int m0    = blockIdx.x << 4;
  const int khalf = lane >> 4;
  int row = m0 + (lane & 15);
  if (row >= M) row = M - 1;                // clamp (branch-free OOB handling)

  const float* arow = A + (size_t)row * K + (khalf << 4);
  const v16h*  bp   = (const v16h*)Bp + (size_t)ntile * KS * 32 + lane;

  v8f c = {};
#pragma unroll
  for (int ks = 0; ks < KS; ++ks) {
    const float4* a0p = (const float4*)(arow + ks * 32);
    const float4* a1p = (const float4*)(arow + ks * 32 + 8);
    float4 f0 = a0p[0], f1 = a0p[1];
    float4 f2 = a1p[0], f3 = a1p[1];
    v16h a;
    a[0]  = (_Float16)f0.x; a[1]  = (_Float16)f0.y; a[2]  = (_Float16)f0.z; a[3]  = (_Float16)f0.w;
    a[4]  = (_Float16)f1.x; a[5]  = (_Float16)f1.y; a[6]  = (_Float16)f1.z; a[7]  = (_Float16)f1.w;
    a[8]  = (_Float16)f2.x; a[9]  = (_Float16)f2.y; a[10] = (_Float16)f2.z; a[11] = (_Float16)f2.w;
    a[12] = (_Float16)f3.x; a[13] = (_Float16)f3.y; a[14] = (_Float16)f3.z; a[15] = (_Float16)f3.w;
    v16h b = bp[ks * 32];
    c = __builtin_amdgcn_wmma_f32_16x16x32_f16(false, a, false, b,
                                               (short)0, c, false, false);
  }

  const int col = n0 + (lane & 15);
  float bv = bias ? bias[col] : 0.0f;
#pragma unroll
  for (int r = 0; r < 8; ++r) {
    int rr = m0 + r + (khalf << 3);
    if (rr < M) {
      float v = c[r] + bv;
      if (act == 1)      v = fmaxf(v, 0.0f);
      else if (act == 2) v = tanhf(v);
      C[(size_t)rr * Ncols + col] = v;
    }
  }
}

// ---------- fills ----------------------------------------------------
__global__ void fill_f32(float* p, float v, int n) {
  int i = blockIdx.x * blockDim.x + threadIdx.x;
  if (i < n) p[i] = v;
}
__global__ void fill_u32(unsigned* p, unsigned v, int n) {
  int i = blockIdx.x * blockDim.x + threadIdx.x;
  if (i < n) p[i] = v;
}

// ---------- per-node dual dot (1 wave/node) ---------------------------
__global__ void dual_dot_kernel(const float* __restrict__ xw,
                                const float* __restrict__ asrc,
                                const float* __restrict__ adst,
                                float* __restrict__ as_out,
                                float* __restrict__ ad_out, int n) {
  int lane = threadIdx.x & 31;
  int node = blockIdx.x * (blockDim.x >> 5) + (threadIdx.x >> 5);
  if (node >= n) return;
  const float4 v = ((const float4*)(xw + (size_t)node * H))[lane];
  const float4 ws = ((const float4*)asrc)[lane];
  const float4 wd = ((const float4*)adst)[lane];
  float s = v.x * ws.x + v.y * ws.y + v.z * ws.z + v.w * ws.w;
  float d = v.x * wd.x + v.y * wd.y + v.z * wd.z + v.w * wd.w;
#pragma unroll
  for (int off = 16; off; off >>= 1) {
    s += __shfl_xor(s, off, 32);
    d += __shfl_xor(d, off, 32);
  }
  if (lane == 0) { as_out[node] = s; ad_out[node] = d; }
}

// ---------- edge softmax passes (edges + N self-loops) ---------------
__device__ __forceinline__ void edge_sd(const int* ei, int t, int E, int& s, int& d) {
  if (t < E) { s = ei[t]; d = ei[E + t]; }
  else       { s = t - E; d = t - E; }
}

__global__ void edge_score_max_kernel(const int* __restrict__ ei,
                                      const float* __restrict__ as_,
                                      const float* __restrict__ ad_,
                                      float* __restrict__ ebuf,
                                      unsigned* __restrict__ nmax,
                                      int E, int n) {
  int t = blockIdx.x * blockDim.x + threadIdx.x;
  if (t >= E + n) return;
  int s, d; edge_sd(ei, t, E, s, d);
  float e = as_[s] + ad_[d];
  e = (e > 0.0f) ? e : 0.2f * e;            // leaky_relu(0.2)
  ebuf[t] = e;
  atomicMax(&nmax[d], f2o(e));
}

__global__ void edge_exp_sum_kernel(const int* __restrict__ ei,
                                    float* __restrict__ ebuf,
                                    const unsigned* __restrict__ nmax,
                                    float* __restrict__ nz,
                                    int E, int n) {
  int t = blockIdx.x * blockDim.x + threadIdx.x;
  if (t >= E + n) return;
  int s, d; edge_sd(ei, t, E, s, d);
  float ex = __expf(ebuf[t] - o2f(nmax[d]));
  ebuf[t] = ex;
  atomicAdd(&nz[d], ex);
}

// 1 wave per edge; 4 floats/lane, f32 global atomics into acc[dst]
__global__ void edge_scatter_kernel(const int* __restrict__ ei,
                                    const float* __restrict__ ebuf,
                                    const float* __restrict__ nz,
                                    const float* __restrict__ xw,
                                    float* __restrict__ acc,
                                    int E, int n) {
  int lane = threadIdx.x & 31;
  int t = blockIdx.x * (blockDim.x >> 5) + (threadIdx.x >> 5);
  if (t >= E + n) return;
  int s, d; edge_sd(ei, t, E, s, d);
  float alpha = ebuf[t] / (nz[d] + 1e-16f);
  const float4 v = ((const float4*)(xw + (size_t)s * H))[lane];
  float* ap = acc + (size_t)d * H + lane * 4;
  atomicAdd(ap + 0, alpha * v.x);
  atomicAdd(ap + 1, alpha * v.y);
  atomicAdd(ap + 2, alpha * v.z);
  atomicAdd(ap + 3, alpha * v.w);
}

__global__ void bias_relu_kernel(float* __restrict__ acc,
                                 const float* __restrict__ b, int n) {
  int i = blockIdx.x * blockDim.x + threadIdx.x;
  if (i >= n * H) return;
  acc[i] = fmaxf(acc[i] + b[i & (H - 1)], 0.0f);
}

// ---------- node score: s[i] = u[i].w2 + b2 (1 wave/node) -------------
__global__ void score_kernel(const float* __restrict__ u,
                             const float* __restrict__ w2,
                             const float* __restrict__ b2,
                             float* __restrict__ sout, int n) {
  int lane = threadIdx.x & 31;
  int i = blockIdx.x * (blockDim.x >> 5) + (threadIdx.x >> 5);
  if (i >= n) return;
  const float4 v = ((const float4*)(u + (size_t)i * H))[lane];
  const float4 w = ((const float4*)w2)[lane];
  float s = v.x * w.x + v.y * w.y + v.z * w.z + v.w * w.w;
#pragma unroll
  for (int off = 16; off; off >>= 1) s += __shfl_xor(s, off, 32);
  if (lane == 0) sout[i] = s + b2[0];
}

// ---------- graph-segment softmax over nodes --------------------------
__global__ void seg_max_kernel(const float* __restrict__ vals,
                               const int* __restrict__ seg,
                               unsigned* __restrict__ smax, int n) {
  int t = blockIdx.x * blockDim.x + threadIdx.x;
  if (t >= n) return;
  atomicMax(&smax[seg[t]], f2o(vals[t]));
}
__global__ void seg_exp_kernel(const float* __restrict__ vals,
                               const int* __restrict__ seg,
                               const unsigned* __restrict__ smax,
                               float* __restrict__ wbuf,
                               float* __restrict__ sz, int n) {
  int t = blockIdx.x * blockDim.x + threadIdx.x;
  if (t >= n) return;
  int g = seg[t];
  float ex = __expf(vals[t] - o2f(smax[g]));
  wbuf[t] = ex;
  atomicAdd(&sz[g], ex);
}
__global__ void seg_scatter_kernel(const int* __restrict__ seg,
                                   const float* __restrict__ wbuf,
                                   const float* __restrict__ sz,
                                   const float* __restrict__ feat,
                                   float* __restrict__ ctx, int n) {
  int lane = threadIdx.x & 31;
  int t = blockIdx.x * (blockDim.x >> 5) + (threadIdx.x >> 5);
  if (t >= n) return;
  int g = seg[t];
  float alpha = wbuf[t] / (sz[g] + 1e-16f);
  const float4 v = ((const float4*)(feat + (size_t)t * H))[lane];
  float* cp = ctx + (size_t)g * H + lane * 4;
  atomicAdd(cp + 0, alpha * v.x);
  atomicAdd(cp + 1, alpha * v.y);
  atomicAdd(cp + 2, alpha * v.z);
  atomicAdd(cp + 3, alpha * v.w);
}

// ---------- GRU gate update (torch r|z|n layout) ----------------------
__global__ void gru_update_kernel(const float* __restrict__ gi,
                                  const float* __restrict__ gh,
                                  float* __restrict__ h, int g_count) {
  int i = blockIdx.x * blockDim.x + threadIdx.x;
  if (i >= g_count * H) return;
  int g = i / H, c = i % H;
  const float* gir = gi + (size_t)g * 3 * H;
  const float* ghr = gh + (size_t)g * 3 * H;
  float r = 1.0f / (1.0f + __expf(-(gir[c] + ghr[c])));
  float z = 1.0f / (1.0f + __expf(-(gir[H + c] + ghr[H + c])));
  float nn = tanhf(gir[2 * H + c] + r * ghr[2 * H + c]);
  h[i] = (1.0f - z) * nn + z * h[i];
}

// ---------- final: sigmoid(o2 . W3 + b3) (1 wave/graph, len 64) -------
__global__ void final_out_kernel(const float* __restrict__ o2,
                                 const float* __restrict__ w3,
                                 const float* __restrict__ b3,
                                 float* __restrict__ out, int g_count) {
  int lane = threadIdx.x & 31;
  int g = blockIdx.x * (blockDim.x >> 5) + (threadIdx.x >> 5);
  if (g >= g_count) return;
  const float* r = o2 + (size_t)g * (H / 2);
  float s = r[lane] * w3[lane] + r[lane + 32] * w3[lane + 32];
#pragma unroll
  for (int off = 16; off; off >>= 1) s += __shfl_xor(s, off, 32);
  if (lane == 0) out[g] = 1.0f / (1.0f + __expf(-(s + b3[0])));
}

// =====================================================================
extern "C" void kernel_launch(void* const* d_in, const int* in_sizes, int n_in,
                              void* d_out, int out_size, void* d_ws, size_t ws_size,
                              hipStream_t stream) {
  (void)n_in; (void)ws_size;
  const float* x      = (const float*)d_in[0];
  const int*   ei     = (const int*)  d_in[1];
  const int*   batch  = (const int*)  d_in[2];
  const float* enc_W  = (const float*)d_in[3];
  const float* enc_b  = (const float*)d_in[4];
  const float* gat_W  = (const float*)d_in[5];
  const float* gat_as = (const float*)d_in[6];
  const float* gat_ad = (const float*)d_in[7];
  const float* gat_b  = (const float*)d_in[8];
  const float* att_W1 = (const float*)d_in[9];
  const float* att_b1 = (const float*)d_in[10];
  const float* att_w2 = (const float*)d_in[11];
  const float* att_b2 = (const float*)d_in[12];
  const float* Wih    = (const float*)d_in[13];
  const float* Whh    = (const float*)d_in[14];
  const float* bih    = (const float*)d_in[15];
  const float* bhh    = (const float*)d_in[16];
  const float* oW1    = (const float*)d_in[17];
  const float* ob1    = (const float*)d_in[18];
  const float* oW2    = (const float*)d_in[19];
  const float* ob2    = (const float*)d_in[20];
  const float* oW3    = (const float*)d_in[21];
  const float* ob3    = (const float*)d_in[22];

  const int N = in_sizes[0] / DIN;
  const int E = in_sizes[1] / 2;
  const int G = out_size;
  const int ETOT = E + N;                      // edges + self loops

  // ---- workspace carve-out (256B aligned slices) ----
  char* wp = (char*)d_ws;
  auto carve = [&](size_t bytes) -> char* {
    char* p = wp;
    wp += (bytes + 255) & ~(size_t)255;
    return p;
  };
  float*     bufA  = (float*)carve((size_t)N * H * 4);
  float*     bufB  = (float*)carve((size_t)N * H * 4);
  float*     bufC  = (float*)carve((size_t)N * H * 4);
  float*     a_s   = (float*)carve((size_t)N * 4);
  float*     a_d   = (float*)carve((size_t)N * 4);
  float*     sbuf  = (float*)carve((size_t)N * 4);
  float*     ebuf  = (float*)carve((size_t)ETOT * 4);
  unsigned*  nmax  = (unsigned*)carve((size_t)N * 4);
  float*     nz    = (float*)carve((size_t)N * 4);
  unsigned*  gmax  = (unsigned*)carve((size_t)G * 4);
  float*     gz    = (float*)carve((size_t)G * 4);
  float*     ctx   = (float*)carve((size_t)G * H * 4);
  float*     gi    = (float*)carve((size_t)G * 3 * H * 4);
  float*     gh    = (float*)carve((size_t)G * 3 * H * 4);
  float*     hst   = (float*)carve((size_t)G * H * 4);
  float*     o1    = (float*)carve((size_t)G * H * 4);
  float*     o2    = (float*)carve((size_t)G * (H / 2) * 4);
  _Float16*  pb    = (_Float16*)carve((size_t)3 * H * H * 2);   // shared packed-B
  _Float16*  pbWhh = (_Float16*)carve((size_t)3 * H * H * 2);   // persistent Whh pack

  const unsigned NEG_INF_ORD = 0x007FFFFFu;    // f2o(-inf)
  auto cdiv = [](int a, int b) { return (a + b - 1) / b; };

  auto packB = [&](const float* B, _Float16* Bp, int K, int Nc, int tB) {
    int total = (Nc >> 4) * (K >> 5) * 256;
    hipLaunchKernelGGL(pack_b_kernel, dim3(cdiv(total, 256)), dim3(256), 0, stream,
                       B, Bp, K, Nc, tB);
  };
  auto gemmP = [&](const float* A, const _Float16* Bp, const float* bias, float* C,
                   int M, int K, int Nc, int act) {
    dim3 grid(cdiv(M, 16), cdiv(Nc, 64));
    if (K == 64)
      hipLaunchKernelGGL(wmma_gemm_packed_kernel<64>, grid, dim3(128), 0, stream,
                         A, Bp, bias, C, M, Nc, act);
    else
      hipLaunchKernelGGL(wmma_gemm_packed_kernel<128>, grid, dim3(128), 0, stream,
                         A, Bp, bias, C, M, Nc, act);
  };
  auto gemm = [&](const float* A, const float* B, const float* bias, float* C,
                  int M, int K, int Nc, int tB, int act) {
    packB(B, pb, K, Nc, tB);
    gemmP(A, pb, bias, C, M, K, Nc, act);
  };

  float* h   = bufA;   // current node features
  float* acc = bufC;   // scatter accumulator (swaps with h each layer)

  // 1) node encoder: h = relu(x @ enc_W + enc_b)
  gemm(x, enc_W, enc_b, h, N, DIN, H, 0, 1);

  // 2) GAT layers
  for (int l = 0; l < NLAYERS; ++l) {
    gemm(h, gat_W + (size_t)l * H * H, nullptr, bufB, N, H, H, 0, 0);   // xw
    hipLaunchKernelGGL(dual_dot_kernel, dim3(cdiv(N, 8)), dim3(256), 0, stream,
                       bufB, gat_as + l * H, gat_ad + l * H, a_s, a_d, N);
    hipLaunchKernelGGL(fill_u32, dim3(cdiv(N, 256)), dim3(256), 0, stream,
                       nmax, NEG_INF_ORD, N);
    hipLaunchKernelGGL(fill_f32, dim3(cdiv(N, 256)), dim3(256), 0, stream,
                       nz, 0.0f, N);
    hipLaunchKernelGGL(fill_f32, dim3(cdiv(N * H, 256)), dim3(256), 0, stream,
                       acc, 0.0f, N * H);
    hipLaunchKernelGGL(edge_score_max_kernel, dim3(cdiv(ETOT, 256)), dim3(256), 0, stream,
                       ei, a_s, a_d, ebuf, nmax, E, N);
    hipLaunchKernelGGL(edge_exp_sum_kernel, dim3(cdiv(ETOT, 256)), dim3(256), 0, stream,
                       ei, ebuf, nmax, nz, E, N);
    hipLaunchKernelGGL(edge_scatter_kernel, dim3(cdiv(ETOT, 8)), dim3(256), 0, stream,
                       ei, ebuf, nz, bufB, acc, E, N);
    hipLaunchKernelGGL(bias_relu_kernel, dim3(cdiv(N * H, 256)), dim3(256), 0, stream,
                       acc, gat_b + l * H, N);
    float* t = h; h = acc; acc = t;   // new features live in old acc
  }

  // 3) attentive readout (s, w, context invariant across the T loop)
  gemm(h, att_W1, att_b1, bufB, N, H, H, 0, 2);                // u = tanh(...)
  hipLaunchKernelGGL(score_kernel, dim3(cdiv(N, 8)), dim3(256), 0, stream,
                     bufB, att_w2, att_b2, sbuf, N);
  hipLaunchKernelGGL(fill_u32, dim3(cdiv(G, 256)), dim3(256), 0, stream,
                     gmax, NEG_INF_ORD, G);
  hipLaunchKernelGGL(fill_f32, dim3(cdiv(G, 256)), dim3(256), 0, stream,
                     gz, 0.0f, G);
  hipLaunchKernelGGL(fill_f32, dim3(cdiv(G * H, 256)), dim3(256), 0, stream,
                     ctx, 0.0f, G * H);
  hipLaunchKernelGGL(seg_max_kernel, dim3(cdiv(N, 256)), dim3(256), 0, stream,
                     sbuf, batch, gmax, N);
  hipLaunchKernelGGL(seg_exp_kernel, dim3(cdiv(N, 256)), dim3(256), 0, stream,
                     sbuf, batch, gmax, ebuf, gz, N);
  hipLaunchKernelGGL(seg_scatter_kernel, dim3(cdiv(N, 8)), dim3(256), 0, stream,
                     batch, ebuf, gz, h, ctx, N);

  // gi = context @ Wih^T + bih (loop-invariant); pack Whh once for the loop
  gemm(ctx, Wih, bih, gi, G, H, 3 * H, 1, 0);
  packB(Whh, pbWhh, H, 3 * H, 1);
  hipLaunchKernelGGL(fill_f32, dim3(cdiv(G * H, 256)), dim3(256), 0, stream,
                     hst, 0.0f, G * H);
  for (int t = 0; t < TSTEPS; ++t) {
    gemmP(hst, pbWhh, bhh, gh, G, H, 3 * H, 0);                // gh = h @ Whh^T + bhh
    hipLaunchKernelGGL(gru_update_kernel, dim3(cdiv(G * H, 256)), dim3(256), 0, stream,
                       gi, gh, hst, G);
  }

  // 4) output MLP
  gemm(hst, oW1, ob1, o1, G, H, H, 0, 1);
  gemm(o1, oW2, ob2, o2, G, H, H / 2, 0, 1);
  hipLaunchKernelGGL(final_out_kernel, dim3(cdiv(G, 8)), dim3(256), 0, stream,
                     o2, oW3, ob3, (float*)d_out, G);
}
